// MultiHeadMLPAttentionModel_71201967833249
// MI455X (gfx1250) — matrix operand order repaired and checked by reading, not compiled, not run on hardware
//
#include <hip/hip_runtime.h>
#include <math.h>

typedef __attribute__((ext_vector_type(16))) _Float16 v16h;
typedef __attribute__((ext_vector_type(8)))  float    v8f;

#define HID    64
#define HEADS  4
#define NPTS   8192
#define NTHR   512
#define NWAVES 16   // wave32: 512 threads = 16 waves

struct Smem {
  float    w[NPTS][HEADS];        // logits, then unnormalized exp weights (128 KB)
  _Float16 w2t[HID * HID];        // enc_w2 transposed [n][k], f16 (8 KB)
  float    encB[HID];             // radar-folded enc L1 bias
  float    eW1x[HID], eW1y[HID];  // pts rows of enc_w1
  float    scB[HEADS][HID];       // radar-folded score L1 bias
  float    sW1x[HEADS][HID], sW1y[HEADS][HID];
  float    scW2[HEADS][HID];
  float    encb2[HID];
  float    ctx[HEADS][HID];       // pooled contexts
  float    red[NWAVES * HEADS];
  float    mh[HEADS], sh[HEADS];
  float    outh[HID];
};

extern "C" __global__ __launch_bounds__(NTHR)
void mh_mlp_attn_fused(const float* __restrict__ radar_xy,
                       const float* __restrict__ radar_dir,
                       const float* __restrict__ pts,
                       const float* __restrict__ enc_w1, const float* __restrict__ enc_b1,
                       const float* __restrict__ enc_w2, const float* __restrict__ enc_b2,
                       const float* __restrict__ sc_w1,  const float* __restrict__ sc_b1,
                       const float* __restrict__ sc_w2,  const float* __restrict__ sc_b2,
                       const float* __restrict__ out_w1, const float* __restrict__ out_b1,
                       const float* __restrict__ out_w2, const float* __restrict__ out_b2,
                       float* __restrict__ out)
{
  extern __shared__ char smem_raw[];
  Smem& S = *reinterpret_cast<Smem*>(smem_raw);
  const int b    = blockIdx.x;
  const int tid  = threadIdx.x;
  const int wave = tid >> 5;
  const int lane = tid & 31;
  const int mrow = lane & 15;   // row (A) / column (B,D) index within 16x16 frag
  const int hi   = lane >> 4;   // upper-half-lane flag

  const float r0 = radar_xy[2*b],  r1 = radar_xy[2*b+1];
  const float r2 = radar_dir[2*b], r3 = radar_dir[2*b+1];

  // ---------------- setup: fold radar into L1 biases, stage weights in LDS
  for (int k = tid; k < HID; k += NTHR) {
    S.encB[k] = fmaf(r0, enc_w1[0*HID+k], fmaf(r1, enc_w1[1*HID+k],
                fmaf(r2, enc_w1[2*HID+k], fmaf(r3, enc_w1[3*HID+k], enc_b1[k]))));
    S.eW1x[k]  = enc_w1[4*HID+k];
    S.eW1y[k]  = enc_w1[5*HID+k];
    S.encb2[k] = enc_b2[k];
  }
  for (int i = tid; i < HEADS*HID; i += NTHR) {
    const int h = i >> 6, k = i & 63;
    const float* w1h = sc_w1 + h*6*HID;
    S.scB[h][k] = fmaf(r0, w1h[0*HID+k], fmaf(r1, w1h[1*HID+k],
                  fmaf(r2, w1h[2*HID+k], fmaf(r3, w1h[3*HID+k], sc_b1[i]))));
    S.sW1x[h][k] = w1h[4*HID+k];
    S.sW1y[h][k] = w1h[5*HID+k];
    S.scW2[h][k] = sc_w2[i];
    S.ctx[h][k]  = 0.f;
  }
  for (int i = tid; i < HID*HID; i += NTHR) {
    const int k = i >> 6, n = i & 63;
    S.w2t[n*HID + k] = (_Float16)enc_w2[i];   // enc_w2 is [k][n] row-major
  }
  __syncthreads();

  // ---------------- pass 1: score logits for all N, per-head max
  // Vectorized: ds_load_b128 for 4 k-values at a time -> DS:VALU ~ 1:4
  const float2* p2 = (const float2*)pts + (size_t)b * NPTS;
  const float sb2[HEADS] = {sc_b2[0], sc_b2[1], sc_b2[2], sc_b2[3]};
  float lmax[HEADS] = {-1e30f, -1e30f, -1e30f, -1e30f};
  for (int i = tid; i < NPTS; i += NTHR) {
    const float2 p = p2[i];
    float acc[HEADS] = {sb2[0], sb2[1], sb2[2], sb2[3]};
    #pragma unroll 4
    for (int kb = 0; kb < HID; kb += 4) {
      #pragma unroll
      for (int h = 0; h < HEADS; ++h) {
        const float4 wx = *(const float4*)&S.sW1x[h][kb];
        const float4 wy = *(const float4*)&S.sW1y[h][kb];
        const float4 cb = *(const float4*)&S.scB[h][kb];
        const float4 w2 = *(const float4*)&S.scW2[h][kb];
        const float v0 = fmaxf(fmaf(p.x, wx.x, fmaf(p.y, wy.x, cb.x)), 0.f);
        const float v1 = fmaxf(fmaf(p.x, wx.y, fmaf(p.y, wy.y, cb.y)), 0.f);
        const float v2 = fmaxf(fmaf(p.x, wx.z, fmaf(p.y, wy.z, cb.z)), 0.f);
        const float v3 = fmaxf(fmaf(p.x, wx.w, fmaf(p.y, wy.w, cb.w)), 0.f);
        acc[h] = fmaf(v0, w2.x, fmaf(v1, w2.y, fmaf(v2, w2.z, fmaf(v3, w2.w, acc[h]))));
      }
    }
    *(float4*)&S.w[i][0] = make_float4(acc[0], acc[1], acc[2], acc[3]);
    #pragma unroll
    for (int h = 0; h < HEADS; ++h) lmax[h] = fmaxf(lmax[h], acc[h]);
  }
  #pragma unroll
  for (int h = 0; h < HEADS; ++h)
    #pragma unroll
    for (int off = 16; off; off >>= 1)
      lmax[h] = fmaxf(lmax[h], __shfl_xor(lmax[h], off, 32));
  if (lane == 0) {
    #pragma unroll
    for (int h = 0; h < HEADS; ++h) S.red[wave*HEADS + h] = lmax[h];
  }
  __syncthreads();
  if (tid < HEADS) {
    float m = S.red[tid];
    for (int w = 1; w < NWAVES; ++w) m = fmaxf(m, S.red[w*HEADS + tid]);
    S.mh[tid] = m;
  }
  __syncthreads();

  // ---------------- pass 1b: exp in-place + denominator
  const float m0h = S.mh[0], m1h = S.mh[1], m2h = S.mh[2], m3h = S.mh[3];
  float lsum[HEADS] = {0.f, 0.f, 0.f, 0.f};
  for (int i = tid; i < NPTS; i += NTHR) {
    float4 l = *(float4*)&S.w[i][0];
    l.x = __expf(l.x - m0h); l.y = __expf(l.y - m1h);
    l.z = __expf(l.z - m2h); l.w = __expf(l.w - m3h);
    *(float4*)&S.w[i][0] = l;
    lsum[0] += l.x; lsum[1] += l.y; lsum[2] += l.z; lsum[3] += l.w;
  }
  #pragma unroll
  for (int h = 0; h < HEADS; ++h)
    #pragma unroll
    for (int off = 16; off; off >>= 1)
      lsum[h] += __shfl_xor(lsum[h], off, 32);
  if (lane == 0) {
    #pragma unroll
    for (int h = 0; h < HEADS; ++h) S.red[wave*HEADS + h] = lsum[h];
  }
  __syncthreads();
  if (tid < HEADS) {
    float s = 0.f;
    for (int w = 0; w < NWAVES; ++w) s += S.red[w*HEADS + tid];
    S.sh[tid] = s;
  }
  __syncthreads();

  // ---------------- pass 2: WMMA enc L2 fused with weighted pooling
  // B fragments of enc_w2 (16-bit B 32x16 layout: lanes 0-15 K=0..7,16..23; 16-31 K=8..15,24..31)
  v16h Bfrag[4][2];
  #pragma unroll
  for (int t = 0; t < 4; ++t) {
    #pragma unroll
    for (int c = 0; c < 2; ++c) {
      const _Float16* src = &S.w2t[(mrow + 16*t)*HID + c*32 + hi*8];
      v16h bf;
      #pragma unroll
      for (int e = 0; e < 16; ++e) bf[e] = src[(e & 7) + ((e >> 3) << 4)];
      Bfrag[t][c] = bf;
    }
  }

  float ctxacc[4][HEADS];
  #pragma unroll
  for (int t = 0; t < 4; ++t)
    #pragma unroll
    for (int h = 0; h < HEADS; ++h) ctxacc[t][h] = 0.f;

  for (int tile = wave; tile < NPTS/16; tile += NWAVES) {
    const int m0 = tile * 16;
    const float2 p = p2[m0 + mrow];
    // A fragments: h1 = relu(px*W1x + py*W1y + encB), ISA 16-bit A 16x32 layout:
    // lanes0-15 VGPRj: K = 2j (+8 if j>=4); lanes16-31: +8.
    v16h Ac[2];
    #pragma unroll
    for (int c = 0; c < 2; ++c) {
      v16h A;
      #pragma unroll
      for (int s = 0; s < 2; ++s) {
        const int kb = c*32 + s*16 + hi*8;
        const float4* wx = (const float4*)&S.eW1x[kb];
        const float4* wy = (const float4*)&S.eW1y[kb];
        const float4* bb = (const float4*)&S.encB[kb];
        #pragma unroll
        for (int q = 0; q < 2; ++q) {
          const float4 x = wx[q], y = wy[q], bv = bb[q];
          const int e = s*8 + q*4;
          A[e+0] = (_Float16)fmaxf(fmaf(p.x, x.x, fmaf(p.y, y.x, bv.x)), 0.f);
          A[e+1] = (_Float16)fmaxf(fmaf(p.x, x.y, fmaf(p.y, y.y, bv.y)), 0.f);
          A[e+2] = (_Float16)fmaxf(fmaf(p.x, x.z, fmaf(p.y, y.z, bv.z)), 0.f);
          A[e+3] = (_Float16)fmaxf(fmaf(p.x, x.w, fmaf(p.y, y.w, bv.w)), 0.f);
        }
      }
      Ac[c] = A;
    }
    // softmax weights for the 8 rows this lane pools (D frag: M = r + hi*8)
    float4 wv[8];
    #pragma unroll
    for (int r = 0; r < 8; ++r)
      wv[r] = *(const float4*)&S.w[m0 + hi*8 + r][0];
    #pragma unroll
    for (int t = 0; t < 4; ++t) {
      v8f cfrag = {};
      cfrag = __builtin_amdgcn_wmma_f32_16x16x32_f16(false, Ac[0], false, Bfrag[t][0],
                                                     (short)0, cfrag, false, false);
      cfrag = __builtin_amdgcn_wmma_f32_16x16x32_f16(false, Ac[1], false, Bfrag[t][1],
                                                     (short)0, cfrag, false, false);
      #pragma unroll
      for (int r = 0; r < 8; ++r) {
        ctxacc[t][0] = fmaf(wv[r].x, cfrag[r], ctxacc[t][0]);
        ctxacc[t][1] = fmaf(wv[r].y, cfrag[r], ctxacc[t][1]);
        ctxacc[t][2] = fmaf(wv[r].z, cfrag[r], ctxacc[t][2]);
        ctxacc[t][3] = fmaf(wv[r].w, cfrag[r], ctxacc[t][3]);
      }
    }
  }
  #pragma unroll
  for (int t = 0; t < 4; ++t)
    #pragma unroll
    for (int h = 0; h < HEADS; ++h)
      atomicAdd(&S.ctx[h][mrow + 16*t], ctxacc[t][h]);   // ds_add_f32
  __syncthreads();

  // finalize: normalize by softmax denominator; enc_b2 contributes exactly once
  for (int i = tid; i < HEADS*HID; i += NTHR) {
    const int h = i >> 6, d = i & 63;
    S.ctx[h][d] = S.ctx[h][d] / S.sh[h] + S.encb2[d];
  }
  __syncthreads();

  // ---------------- output MLP: 256 -> 64 -> 1
  const float* ctxf = &S.ctx[0][0];
  if (tid < HID) {
    float acc = out_b1[tid];
    for (int i = 0; i < HEADS*HID; ++i)
      acc = fmaf(ctxf[i], out_w1[i*HID + tid], acc);
    S.outh[tid] = fmaxf(acc, 0.f) * out_w2[tid];
  }
  __syncthreads();
  if (tid == 0) {
    float s = out_b2[0];
    for (int i = 0; i < HID; ++i) s += S.outh[i];
    out[b] = s;
  }
}

extern "C" void kernel_launch(void* const* d_in, const int* in_sizes, int n_in,
                              void* d_out, int out_size, void* d_ws, size_t ws_size,
                              hipStream_t stream) {
  const float* radar_xy  = (const float*)d_in[0];
  const float* radar_dir = (const float*)d_in[1];
  const float* pts       = (const float*)d_in[2];
  const float* enc_w1    = (const float*)d_in[3];
  const float* enc_b1    = (const float*)d_in[4];
  const float* enc_w2    = (const float*)d_in[5];
  const float* enc_b2    = (const float*)d_in[6];
  const float* sc_w1     = (const float*)d_in[7];
  const float* sc_b1     = (const float*)d_in[8];
  const float* sc_w2     = (const float*)d_in[9];
  const float* sc_b2     = (const float*)d_in[10];
  const float* out_w1    = (const float*)d_in[11];
  const float* out_b1    = (const float*)d_in[12];
  const float* out_w2    = (const float*)d_in[13];
  const float* out_b2    = (const float*)d_in[14];
  float* out = (float*)d_out;

  const int Bn = in_sizes[0] / 2;          // 128
  const size_t shmem = sizeof(Smem);       // ~146 KB (< 320 KB WGP LDS)
  hipFuncSetAttribute((const void*)mh_mlp_attn_fused,
                      hipFuncAttributeMaxDynamicSharedMemorySize, (int)shmem);
  mh_mlp_attn_fused<<<dim3(Bn), dim3(NTHR), shmem, stream>>>(
      radar_xy, radar_dir, pts,
      enc_w1, enc_b1, enc_w2, enc_b2,
      sc_w1, sc_b1, sc_w2, sc_b2,
      out_w1, out_b1, out_w2, out_b2, out);
}
